// Attention_50027779064227
// MI455X (gfx1250) — matrix operand order repaired and checked by reading, not compiled, not run on hardware
//
#include <hip/hip_runtime.h>
#include <hip/hip_bf16.h>

// ---------------------------------------------------------------------------
// Attention scoring: out = softmax_s( v . tanh(concat(hidden, enc) @ W + b) )
//   hidden [64,512] f32, enc [1024,64,1024] f32, W [1536,512], b[512], v[512]
// Decomposed: hproj = hidden @ W_top + b (tiny, s-independent)
//             scores[b,s] = v . tanh(enc[s,b] @ W_bot + hproj[b])
// Main GEMM: (64*1024) x 1024 x 512 bf16 WMMA, f32 accumulate.
// ---------------------------------------------------------------------------

#define BATCH   64
#define SLEN    1024
#define DHID    512        // N of the big GEMM
#define KENC    1024       // K of the big GEMM (2*ENC_HID)

typedef __attribute__((ext_vector_type(16))) __bf16 v16bf;
typedef __attribute__((ext_vector_type(8)))  float  v8f;

__device__ __forceinline__ unsigned f32_to_bf16_bits(float f) {
    unsigned u = __builtin_bit_cast(unsigned, f);
    u += 0x7FFFu + ((u >> 16) & 1u);          // round-to-nearest-even
    return u >> 16;
}
__device__ __forceinline__ unsigned pack_bf16x2(float lo, float hi) {
    return f32_to_bf16_bits(lo) | (f32_to_bf16_bits(hi) << 16);
}

// Hardware V_TANH_F32 on gfx1250 if the builtin exists; else expf-based.
#if __has_builtin(__builtin_amdgcn_tanhf)
__device__ __forceinline__ float fast_tanh(float x) { return __builtin_amdgcn_tanhf(x); }
#elif __has_builtin(__builtin_amdgcn_tanh_f32)
__device__ __forceinline__ float fast_tanh(float x) { return __builtin_amdgcn_tanh_f32(x); }
#else
__device__ __forceinline__ float fast_tanh(float x) {
    float e = __expf(2.0f * x);               // hw v_exp_f32
    return 1.0f - 2.0f / (e + 1.0f);
}
#endif

// ---------------------------------------------------------------------------
// Kernel 1: hproj[b][n] = bias[n] + sum_k hidden[b][k] * W[k][n]   (k < 512)
// ---------------------------------------------------------------------------
__global__ void prep_hproj(const float* __restrict__ hidden,
                           const float* __restrict__ W,
                           const float* __restrict__ bias,
                           float* __restrict__ hproj) {
    int i = blockIdx.x * blockDim.x + threadIdx.x;   // 0 .. 64*512-1
    int b = i >> 9;
    int n = i & 511;
    float acc = bias[n];
    const float* hrow = hidden + b * DHID;
#pragma unroll 8
    for (int k = 0; k < DHID; ++k)
        acc = fmaf(hrow[k], W[k * DHID + n], acc);
    hproj[i] = acc;
}

// ---------------------------------------------------------------------------
// Kernel 2: pre-swizzle W_bot (rows 512..1535 of W) into bf16 B-fragments.
// Fragment f = nc*32 + kc covers K = kc*32..+31, N = nc*16..+15.
// B order used here (32x16, same as A): lanes 0-15 hold K=0..15 (elem e -> K=e,
// packed pairs per VGPR), lanes 16-31 hold K=16..31.  Stored lane-contiguous:
// frag base + lane*16 elems (32 B per lane -> one clean b128 pair).
// ---------------------------------------------------------------------------
__global__ void prep_bswz(const float* __restrict__ W,
                          unsigned short* __restrict__ wsB) {
    int i = blockIdx.x * blockDim.x + threadIdx.x;   // 0 .. 1024*512-1
    int f    = i >> 9;
    int r    = i & 511;
    int lane = r >> 4;
    int e    = r & 15;
    int nc   = f >> 5;
    int kc   = f & 31;
    int k = kc * 32 + ((lane >= 16) ? 16 : 0) + e;
    int n = nc * 16 + (lane & 15);
    wsB[i] = (unsigned short)f32_to_bf16_bits(W[(DHID + k) * DHID + n]);
}

// ---------------------------------------------------------------------------
// Kernel 3: fused GEMM + tanh + dot(v) -> scores.
// grid = (S/32, B); block = 128 threads (4 waves, wave32).
// M-tile = 32 rows (2 A fragments shared via LDS, double-buffered).
// Wave w owns N columns [w*128, w*128+128): 8 B fragments reused across the
// 2 M fragments -> 16 WMMAs per K-iter per wave, 1 global b128 per WMMA.
// B fragments are software-pipelined: the load for tile j+1 issues before
// the two WMMAs of tile j, hiding L2 latency behind WMMA execution.
// ---------------------------------------------------------------------------
__global__ void __launch_bounds__(128)
attn_gemm(const float* __restrict__ enc,            // [S][B][KENC]
          const unsigned short* __restrict__ wsB,   // swizzled bf16 W_bot
          const float* __restrict__ hproj,          // [B][DHID]
          const float* __restrict__ vvec,           // [DHID]
          float* __restrict__ scores) {             // [B][S]
    __shared__ __align__(32) unsigned short lds_a[2][1024];  // 2 x 2KB
    __shared__ float sred[4][32];

    const int b    = blockIdx.y;
    const int s0   = blockIdx.x * 32;
    const int tid  = threadIdx.x;
    const int wv   = tid >> 5;
    const int lane = tid & 31;

    // ---- staging indices: each thread fills 8 consecutive bf16 elements ----
    const int j0     = tid * 8;            // 0..1016 over 2 fragments
    const int frag   = j0 >> 9;            // 0 or 1 (rows s0+frag*16 ..)
    const int r0     = j0 & 511;
    const int lane_s = r0 >> 4;            // fragment lane
    const int e0     = r0 & 15;            // 0 or 8
    const int m      = lane_s & 15;
    // A layout: elem e -> K = e + (lane>=16?16:0); e0..e0+7 is
    // K-contiguous (e0=0 -> K=loff.., e0=8 -> K=8+loff..).
    const int kin = e0 + ((lane_s >= 16) ? 16 : 0);
    const float* arow =
        enc + ((size_t)(s0 + frag * 16 + m) * BATCH + b) * KENC + kin;

    // Per-wave B base for kc=0: fragment f = (wv*8+j)*32 + kc, elem f*512.
    // Stride over j: 32*512 elems; stride over kc: 512 elems.
    const unsigned short* bbase0 =
        wsB + ((size_t)(wv * 8) * 32) * 512 + lane * 16;

    v8f acc0[8] = {};
    v8f acc1[8] = {};

    for (int kc = 0; kc < 32; ++kc) {
        // ---- stage 32x32 A tile (f32 -> bf16, fragment-swizzled) ----
        const int buf = kc & 1;
        const float4 fa = *(const float4*)(arow + kc * 32);
        const float4 fb = *(const float4*)(arow + kc * 32 + 4);
        uint4 p;
        p.x = pack_bf16x2(fa.x, fa.y);
        p.y = pack_bf16x2(fa.z, fa.w);
        p.z = pack_bf16x2(fb.x, fb.y);
        p.w = pack_bf16x2(fb.z, fb.w);
        *(uint4*)&lds_a[buf][j0] = p;

        const unsigned short* bb = bbase0 + (size_t)kc * 512;
        __syncthreads();

        const v16bf a0 = *(const v16bf*)(&lds_a[buf][lane * 16]);
        const v16bf a1 = *(const v16bf*)(&lds_a[buf][512 + lane * 16]);

        // Software-pipelined B fragments.
        v16bf bcur = *(const v16bf*)(bb);
#pragma unroll
        for (int j = 0; j < 8; ++j) {
            v16bf bnext;
            if (j < 7)
                bnext = *(const v16bf*)(bb + (size_t)(j + 1) * (32 * 512));
            acc0[j] = __builtin_amdgcn_wmma_f32_16x16x32_bf16(
                false, a0, false, bcur, (short)0, acc0[j], false, false);
            acc1[j] = __builtin_amdgcn_wmma_f32_16x16x32_bf16(
                false, a1, false, bcur, (short)0, acc1[j], false, false);
            if (j < 7) bcur = bnext;
        }
    }

    // ---- epilogue: +hproj, hw tanh, * v, reduce over N ----
    // C/D layout: lane holds N=(lane&15); VGPR r holds M=r (lanes 0-15) or
    // M=r+8 (lanes 16-31).
    float sp0[8] = {0.f, 0.f, 0.f, 0.f, 0.f, 0.f, 0.f, 0.f};
    float sp1[8] = {0.f, 0.f, 0.f, 0.f, 0.f, 0.f, 0.f, 0.f};
#pragma unroll
    for (int j = 0; j < 8; ++j) {
        const int n    = (wv * 8 + j) * 16 + (lane & 15);
        const float hp = hproj[b * DHID + n];
        const float vn = vvec[n];
#pragma unroll
        for (int r = 0; r < 8; ++r) {
            sp0[r] += fast_tanh(acc0[j][r] + hp) * vn;
            sp1[r] += fast_tanh(acc1[j][r] + hp) * vn;
        }
    }
    // butterfly reduce across each 16-lane half (wave32 semantics)
#pragma unroll
    for (int r = 0; r < 8; ++r) {
        float x = sp0[r], y = sp1[r];
        x += __shfl_xor(x, 1, 32);  y += __shfl_xor(y, 1, 32);
        x += __shfl_xor(x, 2, 32);  y += __shfl_xor(y, 2, 32);
        x += __shfl_xor(x, 4, 32);  y += __shfl_xor(y, 4, 32);
        x += __shfl_xor(x, 8, 32);  y += __shfl_xor(y, 8, 32);
        sp0[r] = x;  sp1[r] = y;
    }
    if (lane == 0) {
#pragma unroll
        for (int r = 0; r < 8; ++r) {
            sred[wv][r]      = sp0[r];
            sred[wv][16 + r] = sp1[r];
        }
    } else if (lane == 16) {
#pragma unroll
        for (int r = 0; r < 8; ++r) {
            sred[wv][8 + r]  = sp0[r];
            sred[wv][24 + r] = sp1[r];
        }
    }
    __syncthreads();
    if (tid < 32) {
        float tot = sred[0][tid] + sred[1][tid] + sred[2][tid] + sred[3][tid];
        scores[b * SLEN + s0 + tid] = tot;
    }
}

// ---------------------------------------------------------------------------
// Kernel 4: row-wise softmax over S=1024, one block (256 thr) per batch row.
// ---------------------------------------------------------------------------
__global__ void softmax_rows(const float* __restrict__ scores,
                             float* __restrict__ out) {
    __shared__ float red[256];
    const int b = blockIdx.x;
    const int t = threadIdx.x;
    const float* row = scores + b * SLEN;

    float l0 = row[t], l1 = row[t + 256], l2 = row[t + 512], l3 = row[t + 768];
    float mx = fmaxf(fmaxf(l0, l1), fmaxf(l2, l3));
    red[t] = mx;
    __syncthreads();
    for (int s = 128; s > 0; s >>= 1) {
        if (t < s) red[t] = fmaxf(red[t], red[t + s]);
        __syncthreads();
    }
    mx = red[0];
    __syncthreads();

    float e0 = expf(l0 - mx), e1 = expf(l1 - mx);
    float e2 = expf(l2 - mx), e3 = expf(l3 - mx);
    red[t] = e0 + e1 + e2 + e3;
    __syncthreads();
    for (int s = 128; s > 0; s >>= 1) {
        if (t < s) red[t] += red[t + s];
        __syncthreads();
    }
    const float inv = 1.0f / red[0];
    float* orow = out + b * SLEN;
    orow[t]       = e0 * inv;
    orow[t + 256] = e1 * inv;
    orow[t + 512] = e2 * inv;
    orow[t + 768] = e3 * inv;
}

// ---------------------------------------------------------------------------
extern "C" void kernel_launch(void* const* d_in, const int* in_sizes, int n_in,
                              void* d_out, int out_size, void* d_ws, size_t ws_size,
                              hipStream_t stream) {
    const float* hidden = (const float*)d_in[0];
    const float* enc    = (const float*)d_in[1];
    const float* W      = (const float*)d_in[2];
    const float* bias   = (const float*)d_in[3];
    const float* vvec   = (const float*)d_in[4];
    float* out = (float*)d_out;

    char* ws = (char*)d_ws;
    float*          hproj  = (float*)ws;                               // 128 KB
    unsigned short* wsB    = (unsigned short*)(ws + BATCH * DHID * 4); // 1 MB
    float*          scores = (float*)(ws + BATCH * DHID * 4
                                         + (size_t)KENC * DHID * 2);   // 256 KB

    prep_hproj<<<(BATCH * DHID) / 256, 256, 0, stream>>>(hidden, W, bias, hproj);
    prep_bswz<<<(KENC * DHID) / 256, 256, 0, stream>>>(W, wsB);
    attn_gemm<<<dim3(SLEN / 32, BATCH), 128, 0, stream>>>(enc, wsB, hproj, vvec, scores);
    softmax_rows<<<BATCH, 256, 0, stream>>>(scores, out);
}
